// HeteroGraphSAGE_43327630082179
// MI455X (gfx1250) — matrix operand order repaired and checked by reading, not compiled, not run on hardware
//
#include <hip/hip_runtime.h>
#include <stdint.h>

#define NN 20000
#define DD 16
#define EE 2
#define FF 128
#define G4 512
#define NT 16
#define SMEM_BYTES 309248

typedef __bf16 v16bf __attribute__((ext_vector_type(16)));
typedef __bf16 v8bf  __attribute__((ext_vector_type(8)));
typedef float  v8f   __attribute__((ext_vector_type(8)));

__device__ __forceinline__ __bf16 f2bf(float f) {
  union { float f; uint32_t u; } v; v.f = f;
  uint32_t r = (v.u + 0x7FFFu + ((v.u >> 16) & 1u)) >> 16;
  union { uint16_t s; __bf16 b; } o; o.s = (uint16_t)r; return o.b;
}
__device__ __forceinline__ float bf2f(__bf16 b) {
  union { __bf16 b; uint16_t s; } i; i.b = b;
  union { uint32_t u; float f; } o; o.u = ((uint32_t)i.s) << 16; return o.f;
}
__device__ __forceinline__ float sigm(float x) { return 1.0f / (1.0f + __expf(-x)); }
__device__ __forceinline__ float tanh_fast(float x) {
  x = fminf(fmaxf(x, -20.0f), 20.0f);
  float e = __expf(2.0f * x);
  return (e - 1.0f) / (e + 1.0f);
}

// ---- CDNA5 async global->LDS copy (tracked by ASYNCcnt) ----
__device__ __forceinline__ void async_ld_b128(void* lds_ptr, const void* gptr) {
  uint32_t la = (uint32_t)(uintptr_t)lds_ptr;           // low 32 bits of generic LDS addr == LDS offset
  uint64_t ga = (uint64_t)(uintptr_t)gptr;
  asm volatile("global_load_async_to_lds_b128 %0, %1, off"
               :: "v"(la), "v"(ga) : "memory");
}
__device__ __forceinline__ void wait_async0() {
  asm volatile("s_wait_asynccnt 0" ::: "memory");
}

// ---- WMMA fragment helpers (wave32, 16x16x32 bf16, fp32 accum) ----
// A: 16x32 (M x K), row-major tile [16][128] bf16: lane m=lane&15,
//    K base = k0 + 8*(lane>=16); elems 0..7 = K..K+7, elems 8..15 = K+16..K+23.
__device__ __forceinline__ v16bf load_afrag(const __bf16* base, int lane, int k0) {
  int m = lane & 15;
  int kb = k0 + ((lane >> 4) << 3);
  const __bf16* p = base + m * FF + kb;
  v8bf lo = *(const v8bf*)p;
  v8bf hi = *(const v8bf*)(p + 16);
  v16bf a;
#pragma unroll
  for (int i = 0; i < 8; ++i) { a[i] = lo[i]; a[i + 8] = hi[i]; }
  return a;
}
// B: 32x16 (K x N) from W^T stored [ncol][128]: lane n=ncb+(lane&15),
//    K base = k0 + 8*(lane>=16); elems 0..7 = K..K+7, elems 8..15 = K+16..K+23.
__device__ __forceinline__ v16bf load_bfrag(const __bf16* wT, int lane, int ncb, int k0) {
  int n = ncb + (lane & 15);
  int ks = k0 + ((lane >> 4) << 3);
  const __bf16* p = wT + n * FF + ks;
  v8bf lo = *(const v8bf*)p;
  v8bf hi = *(const v8bf*)(p + 16);
  v16bf b;
#pragma unroll
  for (int i = 0; i < 8; ++i) { b[i] = lo[i]; b[i + 8] = hi[i]; }
  return b;
}
__device__ __forceinline__ v8f wmma_bf16(v16bf a, v16bf b, v8f c) {
  return __builtin_amdgcn_wmma_f32_16x16x32_bf16(false, a, false, b, (short)0, c, false, false);
}

// ---------------- prep kernels ----------------
__global__ void convert_bf16_kernel(const float* __restrict__ in, __bf16* __restrict__ out, int n) {
  int i = blockIdx.x * 256 + threadIdx.x;
  if (i < n) out[i] = f2bf(in[i]);
}

// in: fp32 [B][R][C]  ->  out: bf16 [B][C][R]
__global__ void transpose_bf16_kernel(const float* __restrict__ in, __bf16* __restrict__ out,
                                      int B, int R, int C) {
  long i = (long)blockIdx.x * 256 + threadIdx.x;
  long tot = (long)B * R * C;
  if (i >= tot) return;
  int b = (int)(i / ((long)R * C));
  int rem = (int)(i % ((long)R * C));
  int c = rem / R;
  int r = rem % R;
  out[i] = f2bf(in[((long)b * R + r) * C + c]);
}

// ---------------- main fused SAGE-LSTM layer ----------------
__global__ void __launch_bounds__(256, 1)
sage_layer_kernel(const __bf16* __restrict__ src,   // [N][F] bf16 node features
                  const int* __restrict__ nbr,      // [E][N][D]
                  const int* __restrict__ deg,      // [E][N]
                  const __bf16* __restrict__ wihT,  // [E][4F][F] bf16 (transposed)
                  const __bf16* __restrict__ whhT,  // [E][4F][F]
                  const float* __restrict__ bih,    // [E][4F]
                  const float* __restrict__ bhh,    // [E][4F]
                  const __bf16* __restrict__ wsT,   // [E][F][F] (layer1)
                  const __bf16* __restrict__ wnT,   // [E][F][F] (layer1)
                  const float* __restrict__ bout,   // [E][F]    (layer1)
                  __bf16* __restrict__ hout,        // [N][F]    (layer1)
                  __bf16* __restrict__ hn2,         // [E][N][F] (layer2)
                  int layer) {
  extern __shared__ char smem[];
  __bf16* wih_l   = (__bf16*)(smem);            // 512*128 bf16 = 128KB
  __bf16* whh_l   = (__bf16*)(smem + 131072);   // 128KB
  float*  gates_l = (float*)(smem + 262144);    // 16*512 f32 = 32KB
  __bf16* xt_l    = (__bf16*)(smem + 294912);   // 16*128 bf16 = 4KB
  __bf16* hcur_l  = (__bf16*)(smem + 299008);   // 4KB
  __bf16* hin_l   = (__bf16*)(smem + 303104);   // 4KB
  float*  bias_l  = (float*)(smem + 307200);    // 512 f32 = 2KB

  const int t = threadIdx.x;
  const int w = t >> 5, lane = t & 31;
  const int node = t >> 4, ch = t & 15;   // cell/gather mapping: 16 threads per node
  const int n0 = blockIdx.x * NT;

  v8f acc2;
#pragma unroll
  for (int r = 0; r < 8; ++r) acc2[r] = 0.0f;

  if (layer == 1) {  // stage self features for the combine GEMM
    async_ld_b128(hin_l + node * FF + ch * 8, src + (size_t)(n0 + node) * FF + ch * 8);
  }

  for (int e = 0; e < EE; ++e) {
    // ---- stage recurrent weights into LDS (async DMA path) ----
    const __bf16* gih = wihT + (size_t)e * G4 * FF;
    const __bf16* ghh = whhT + (size_t)e * G4 * FF;
    for (int i = 0; i < 32; ++i) {
      int c = i * 256 + t;  // 8192 chunks of 16B per matrix
      async_ld_b128(wih_l + c * 8, gih + c * 8);
      async_ld_b128(whh_l + c * 8, ghh + c * 8);
    }
    bias_l[t]       = bih[e * G4 + t]       + bhh[e * G4 + t];
    bias_l[t + 256] = bih[e * G4 + t + 256] + bhh[e * G4 + t + 256];

    float creg[8], hreg[8];
#pragma unroll
    for (int j = 0; j < 8; ++j) {
      creg[j] = 0.0f; hreg[j] = 0.0f;
      hcur_l[node * FF + ch + 16 * j] = f2bf(0.0f);
    }
    const int dn = deg[e * NN + n0 + node];
    wait_async0();
    __syncthreads();

    for (int step = 0; step < DD; ++step) {
      // ---- gather neighbor row for this step directly into LDS ----
      int idx = nbr[((size_t)e * NN + n0 + node) * DD + step];
      async_ld_b128(xt_l + node * FF + ch * 8, src + (size_t)idx * FF + ch * 8);
      wait_async0();
      __syncthreads();

      // ---- gates[16,512] = x_t@Wih + h@Whh + (bih+bhh) via WMMA ----
      v16bf ax[4], ah[4];
#pragma unroll
      for (int k = 0; k < 4; ++k) {
        ax[k] = load_afrag(xt_l, lane, k * 32);
        ah[k] = load_afrag(hcur_l, lane, k * 32);
      }
#pragma unroll
      for (int ti = 0; ti < 4; ++ti) {
        int ncb = w * 64 + ti * 16;
        int col = ncb + (lane & 15);
        float bv = bias_l[col];
        v8f acc;
#pragma unroll
        for (int r = 0; r < 8; ++r) acc[r] = bv;
#pragma unroll
        for (int k = 0; k < 4; ++k) {
          acc = wmma_bf16(ax[k], load_bfrag(wih_l, lane, ncb, k * 32), acc);
          acc = wmma_bf16(ah[k], load_bfrag(whh_l, lane, ncb, k * 32), acc);
        }
        int ro = (lane >> 4) << 3;
#pragma unroll
        for (int r = 0; r < 8; ++r) gates_l[(ro + r) * G4 + col] = acc[r];
      }
      __syncthreads();

      // ---- masked LSTM cell (i,f,g,o order), h back to LDS as bf16 ----
      bool msk = step < dn;
      const float* g = gates_l + node * G4;
#pragma unroll
      for (int j = 0; j < 8; ++j) {
        int u = ch + 16 * j;
        float ig = g[u], fg = g[FF + u], gg = g[2 * FF + u], og = g[3 * FF + u];
        float cn = sigm(fg) * creg[j] + sigm(ig) * tanh_fast(gg);
        float hn = sigm(og) * tanh_fast(cn);
        if (msk) { creg[j] = cn; hreg[j] = hn; }
        hcur_l[node * FF + u] = f2bf(hreg[j]);
      }
      // next-iteration wait+barrier orders hcur/gates reuse
    }
    __syncthreads();  // final h_neigh visible in hcur_l

    if (layer == 1) {
      // out += h@Wself[e] + h_neigh@Wneigh[e]; each wave owns one 16-col tile
      const __bf16* gs = wsT + (size_t)e * FF * FF;
      const __bf16* gn = wnT + (size_t)e * FF * FF;
      int ncb = w * 16;
#pragma unroll
      for (int k = 0; k < 4; ++k) {
        acc2 = wmma_bf16(load_afrag(hin_l, lane, k * 32), load_bfrag(gs, lane, ncb, k * 32), acc2);
        acc2 = wmma_bf16(load_afrag(hcur_l, lane, k * 32), load_bfrag(gn, lane, ncb, k * 32), acc2);
      }
    } else {
      // store per-edge-type LSTM output for the head kernel
      *(uint4*)(hn2 + ((size_t)e * NN + n0 + node) * FF + ch * 8) =
          *(const uint4*)(hcur_l + node * FF + ch * 8);
    }
    __syncthreads();
  }

  if (layer == 1) {
    int col = w * 16 + (lane & 15);
    int ro = (lane >> 4) << 3;
#pragma unroll
    for (int r = 0; r < 8; ++r) gates_l[(ro + r) * FF + col] = acc2[r];
    __syncthreads();
#pragma unroll
    for (int j = 0; j < 8; ++j) {
      int u = ch + 16 * j;
      float v = gates_l[node * FF + u] + bout[u] + bout[FF + u];
      v = 0.5f * v;
      v = v > 0.0f ? v : 0.0f;
      hout[(size_t)(n0 + node) * FF + u] = f2bf(v);
    }
  }
}

// ---------------- head: F->2 projection, logits + softmax ----------------
__global__ void head_kernel(const __bf16* __restrict__ h1,   // [N][F]
                            const __bf16* __restrict__ hn2,  // [E][N][F]
                            const float* __restrict__ Ws,    // [E][F][2]
                            const float* __restrict__ Wn,    // [E][F][2]
                            const float* __restrict__ b2,    // [E][2]
                            float* __restrict__ out) {
  int n = blockIdx.x * 256 + threadIdx.x;
  if (n >= NN) return;
  float l0 = 0.0f, l1 = 0.0f;
  for (int e = 0; e < EE; ++e) {
    const float* ws = Ws + e * FF * 2;
    const float* wn = Wn + e * FF * 2;
#pragma unroll 4
    for (int k = 0; k < FF; ++k) {
      float hv = bf2f(h1[(size_t)n * FF + k]);
      float gv = bf2f(hn2[((size_t)e * NN + n) * FF + k]);
      l0 += hv * ws[k * 2] + gv * wn[k * 2];
      l1 += hv * ws[k * 2 + 1] + gv * wn[k * 2 + 1];
    }
    l0 += b2[e * 2];
    l1 += b2[e * 2 + 1];
  }
  l0 *= 0.5f; l1 *= 0.5f;
  out[(size_t)n * 2] = l0;
  out[(size_t)n * 2 + 1] = l1;
  float m = fmaxf(l0, l1);
  float e0 = __expf(l0 - m), e1 = __expf(l1 - m);
  float inv = 1.0f / (e0 + e1);
  out[2 * NN + (size_t)n * 2] = e0 * inv;
  out[2 * NN + (size_t)n * 2 + 1] = e1 * inv;
}

extern "C" void kernel_launch(void* const* d_in, const int* in_sizes, int n_in,
                              void* d_out, int out_size, void* d_ws, size_t ws_size,
                              hipStream_t stream) {
  (void)in_sizes; (void)n_in; (void)out_size; (void)ws_size;
  const float* x        = (const float*)d_in[0];
  const int*   nbr      = (const int*)d_in[1];
  const int*   deg      = (const int*)d_in[2];
  const float* l1_Wih   = (const float*)d_in[3];
  const float* l1_Whh   = (const float*)d_in[4];
  const float* l1_bih   = (const float*)d_in[5];
  const float* l1_bhh   = (const float*)d_in[6];
  const float* l1_Wself = (const float*)d_in[7];
  const float* l1_Wneigh= (const float*)d_in[8];
  const float* l1_b     = (const float*)d_in[9];
  const float* l2_Wih   = (const float*)d_in[10];
  const float* l2_Whh   = (const float*)d_in[11];
  const float* l2_bih   = (const float*)d_in[12];
  const float* l2_bhh   = (const float*)d_in[13];
  const float* l2_Wself = (const float*)d_in[14];
  const float* l2_Wneigh= (const float*)d_in[15];
  const float* l2_b     = (const float*)d_in[16];

  char* ws = (char*)d_ws;
  __bf16* xbf   = (__bf16*)(ws + 0);          // 5,120,000 B
  __bf16* h1bf  = (__bf16*)(ws + 5120000);    // 5,120,000 B
  __bf16* w1ihT = (__bf16*)(ws + 10240000);   // 262,144 B
  __bf16* w1hhT = (__bf16*)(ws + 10502144);
  __bf16* w2ihT = (__bf16*)(ws + 10764288);
  __bf16* w2hhT = (__bf16*)(ws + 11026432);
  __bf16* w1sT  = (__bf16*)(ws + 11288576);   // 65,536 B
  __bf16* w1nT  = (__bf16*)(ws + 11354112);
  __bf16* hn2   = (__bf16*)(ws + 11419648);   // 10,240,000 B

  convert_bf16_kernel<<<(NN * FF) / 256, 256, 0, stream>>>(x, xbf, NN * FF);
  transpose_bf16_kernel<<<(EE * FF * G4 + 255) / 256, 256, 0, stream>>>(l1_Wih, w1ihT, EE, FF, G4);
  transpose_bf16_kernel<<<(EE * FF * G4 + 255) / 256, 256, 0, stream>>>(l1_Whh, w1hhT, EE, FF, G4);
  transpose_bf16_kernel<<<(EE * FF * G4 + 255) / 256, 256, 0, stream>>>(l2_Wih, w2ihT, EE, FF, G4);
  transpose_bf16_kernel<<<(EE * FF * G4 + 255) / 256, 256, 0, stream>>>(l2_Whh, w2hhT, EE, FF, G4);
  transpose_bf16_kernel<<<(EE * FF * FF + 255) / 256, 256, 0, stream>>>(l1_Wself, w1sT, EE, FF, FF);
  transpose_bf16_kernel<<<(EE * FF * FF + 255) / 256, 256, 0, stream>>>(l1_Wneigh, w1nT, EE, FF, FF);

  sage_layer_kernel<<<NN / NT, 256, SMEM_BYTES, stream>>>(
      xbf, nbr, deg, w1ihT, w1hhT, l1_bih, l1_bhh, w1sT, w1nT, l1_b, h1bf, nullptr, 1);
  sage_layer_kernel<<<NN / NT, 256, SMEM_BYTES, stream>>>(
      h1bf, nbr, deg, w2ihT, w2hhT, l2_bih, l2_bhh, nullptr, nullptr, nullptr, nullptr, hn2, 2);

  head_kernel<<<(NN + 255) / 256, 256, 0, stream>>>(h1bf, hn2, l2_Wself, l2_Wneigh, l2_b,
                                                    (float*)d_out);
}
